// SocialLSTM_90477781057851
// MI455X (gfx1250) — matrix operand order repaired and checked by reading, not compiled, not run on hardware
//
#include <hip/hip_runtime.h>
#include <hip/hip_bf16.h>

typedef __attribute__((ext_vector_type(16))) _Float16 v16h;
typedef __attribute__((ext_vector_type(8)))  float    v8f;

#define NAGENT 512
#define TOBS   8
#define HDIM   128
#define EDIM   64
#define PDIM   128
#define GG     16      // G*G
#define KPOOL  2048    // G*G*H
#define XDIM   320     // E + P + H (concat [emb, soc, h])
#define NGATE  512     // 4*H

union FragH { v16h v; unsigned u[8]; };

__device__ __forceinline__ float sigmoidf_(float x) { return 1.0f / (1.0f + __expf(-x)); }

// ---------------------------------------------------------------- impute ----
__global__ void impute_kernel(const float* __restrict__ observed, float* __restrict__ obs) {
  int n = blockIdx.x * blockDim.x + threadIdx.x;
  if (n >= NAGENT) return;
  bool fin[TOBS];
  int first = -1;
  bool anyv = false;
#pragma unroll
  for (int t = 0; t < TOBS; t++) {
    float x = observed[(t * NAGENT + n) * 2 + 0];
    float y = observed[(t * NAGENT + n) * 2 + 1];
    bool f = __builtin_isfinite(x) && __builtin_isfinite(y);
    fin[t] = f;
    if (f && first < 0) first = t;
    anyv = anyv || f;
  }
  int last = -1;
#pragma unroll
  for (int t = 0; t < TOBS; t++) {
    if (fin[t]) last = t;
    int take = (last >= 0) ? last : first;
    float ox = 0.0f, oy = 0.0f;
    if (anyv) {
      ox = observed[(take * NAGENT + n) * 2 + 0];
      oy = observed[(take * NAGENT + n) * 2 + 1];
    }
    obs[(t * NAGENT + n) * 2 + 0] = ox;
    obs[(t * NAGENT + n) * 2 + 1] = oy;
  }
}

// ------------------------------------------------- weight prep + state init ----
__global__ void prep_kernel(const float* __restrict__ Wpool,
                            const float* __restrict__ WihE, const float* __restrict__ WhhE,
                            const float* __restrict__ WihD, const float* __restrict__ WhhD,
                            const float* __restrict__ bihE, const float* __restrict__ bhhE,
                            const float* __restrict__ bihD, const float* __restrict__ bhhD,
                            _Float16* __restrict__ Wpool16,
                            _Float16* __restrict__ WcatE, _Float16* __restrict__ WcatD,
                            float* __restrict__ bgE, float* __restrict__ bgD,
                            float* __restrict__ hs, float* __restrict__ cs,
                            _Float16* __restrict__ xh) {
  int stride = gridDim.x * blockDim.x;
  for (int i = blockIdx.x * blockDim.x + threadIdx.x; i < PDIM * KPOOL; i += stride) {
    Wpool16[i] = (_Float16)Wpool[i];
    if (i < NGATE * XDIM) {
      int r = i / XDIM, c = i % XDIM;
      float ve = (c < EDIM + PDIM) ? WihE[r * (EDIM + PDIM) + c] : WhhE[r * HDIM + (c - EDIM - PDIM)];
      float vd = (c < EDIM + PDIM) ? WihD[r * (EDIM + PDIM) + c] : WhhD[r * HDIM + (c - EDIM - PDIM)];
      WcatE[i] = (_Float16)ve;
      WcatD[i] = (_Float16)vd;
    }
    if (i < NGATE) { bgE[i] = bihE[i] + bhhE[i]; bgD[i] = bihD[i] + bhhD[i]; }
    if (i < NAGENT * HDIM) {
      hs[i] = 0.0f; cs[i] = 0.0f;
      int r = i / HDIM, c = i % HDIM;
      xh[(size_t)r * XDIM + (EDIM + PDIM) + c] = (_Float16)0.0f;
    }
  }
}

// ------------------------------------------------- social grid max-pooling ----
__global__ __launch_bounds__(HDIM) void pool_kernel(const float* __restrict__ pos,
                                                    const float* __restrict__ hs,
                                                    _Float16* __restrict__ pooled) {
  __shared__ float spos[NAGENT * 2];
  int i = blockIdx.x;
  int h = threadIdx.x;
  for (int j = h; j < NAGENT * 2; j += HDIM) spos[j] = pos[j];
  __syncthreads();
  float px = spos[2 * i], py = spos[2 * i + 1];
  float best[GG];
#pragma unroll
  for (int c = 0; c < GG; c++) best[c] = -__builtin_inff();
  for (int j = 0; j < NAGENT; j++) {
    float rx = spos[2 * j] - px;
    float ry = spos[2 * j + 1] - py;
    bool ok = (fabsf(rx) <= 1.0f) && (fabsf(ry) <= 1.0f) && (j != i);
    int gx = min(max((int)floorf((rx + 1.0f) * 2.0f), 0), 3);
    int gy = min(max((int)floorf((ry + 1.0f) * 2.0f), 0), 3);
    int cid = gx * 4 + gy;
    float hv = hs[j * HDIM + h];
#pragma unroll
    for (int c = 0; c < GG; c++)
      if (ok && (cid == c)) best[c] = fmaxf(best[c], hv);
  }
#pragma unroll
  for (int c = 0; c < GG; c++) {
    float v = (best[c] == -__builtin_inff()) ? 0.0f : best[c];
    pooled[(size_t)i * KPOOL + c * HDIM + h] = (_Float16)v;
  }
}

// ------------------------------------------------- velocity embedding pack ----
__global__ void emb_kernel(const float* __restrict__ pos, const float* __restrict__ prev,
                           const float* __restrict__ Wpos, const float* __restrict__ bpos,
                           _Float16* __restrict__ xh) {
  int idx = blockIdx.x * blockDim.x + threadIdx.x;
  if (idx >= NAGENT * EDIM) return;
  int n = idx / EDIM, e = idx % EDIM;
  float vx = pos[2 * n] - prev[2 * n];
  float vy = pos[2 * n + 1] - prev[2 * n + 1];
  float v = fmaxf(0.0f, Wpos[2 * e] * vx + Wpos[2 * e + 1] * vy + bpos[e]);
  xh[(size_t)n * XDIM + e] = (_Float16)v;
}

// ------------------- WMMA GEMM #1: soc = relu(pooled[512,2048] @ Wpool^T + b) ----
// one wave = one 16x16 tile of [512 x 128]; K = 2048, 64 wmma iterations
__global__ __launch_bounds__(128) void socgemm_kernel(const _Float16* __restrict__ A,
                                                      const _Float16* __restrict__ W,   // [128,2048] == B^T
                                                      const float* __restrict__ bpool,
                                                      _Float16* __restrict__ xh) {
  int wid  = (blockIdx.x * blockDim.x + threadIdx.x) >> 5;   // 0..255
  int lane = threadIdx.x & 31;
  int tm = wid >> 3;          // 0..31 (agent tile)
  int tn = wid & 7;           // 0..7  (P tile)
  int hi = lane >> 4;
  int l15 = lane & 15;
  const _Float16* Arow = A + (size_t)(tm * 16 + l15) * KPOOL;
  const _Float16* Brow = W + (size_t)(tn * 16 + l15) * KPOOL;
  v8f acc = {};
  for (int kb = 0; kb < KPOOL; kb += 32) {
    FragH a, b;
#pragma unroll
    for (int p = 0; p < 8; p++) {
      int ka  = kb + hi * 16 + 2 * p;                                              // A layout (contiguous k, same as B)
      int kbb = kb + hi * 16 + 2 * p;                                              // B layout
      a.u[p] = *reinterpret_cast<const unsigned*>(Arow + ka);
      b.u[p] = *reinterpret_cast<const unsigned*>(Brow + kbb);
    }
    acc = __builtin_amdgcn_wmma_f32_16x16x32_f16(false, a.v, false, b.v, (short)0, acc, false, false);
  }
  int col = tn * 16 + l15;
  float bb = bpool[col];
#pragma unroll
  for (int v = 0; v < 8; v++) {
    int row = tm * 16 + v + hi * 8;
    float s = fmaxf(0.0f, acc[v] + bb);
    xh[(size_t)row * XDIM + EDIM + col] = (_Float16)s;   // soc goes to xh[:,64:192]
  }
}

// ------------- WMMA GEMM #2: gates = xh[512,320] @ Wcat^T + (b_ih + b_hh) ----
__global__ __launch_bounds__(256) void gatesgemm_kernel(const _Float16* __restrict__ A,
                                                        const _Float16* __restrict__ W,   // [512,320] == B^T
                                                        const float* __restrict__ bg,
                                                        float* __restrict__ gates) {
  int wid  = (blockIdx.x * blockDim.x + threadIdx.x) >> 5;   // 0..1023
  int lane = threadIdx.x & 31;
  int tm = wid >> 5;          // 0..31 (agent tile)
  int tn = wid & 31;          // 0..31 (gate tile)
  int hi = lane >> 4;
  int l15 = lane & 15;
  const _Float16* Arow = A + (size_t)(tm * 16 + l15) * XDIM;
  const _Float16* Brow = W + (size_t)(tn * 16 + l15) * XDIM;
  v8f acc = {};
#pragma unroll
  for (int kb = 0; kb < XDIM; kb += 32) {
    FragH a, b;
#pragma unroll
    for (int p = 0; p < 8; p++) {
      int ka  = kb + hi * 16 + 2 * p;
      int kbb = kb + hi * 16 + 2 * p;
      a.u[p] = *reinterpret_cast<const unsigned*>(Arow + ka);
      b.u[p] = *reinterpret_cast<const unsigned*>(Brow + kbb);
    }
    acc = __builtin_amdgcn_wmma_f32_16x16x32_f16(false, a.v, false, b.v, (short)0, acc, false, false);
  }
  int col = tn * 16 + l15;
  float bb = bg[col];
#pragma unroll
  for (int v = 0; v < 8; v++) {
    int row = tm * 16 + v + hi * 8;
    gates[(size_t)row * NGATE + col] = acc[v] + bb;
  }
}

// ---------------------------------------- LSTM elementwise + output head ----
__global__ __launch_bounds__(HDIM) void lstm_kernel(const float* __restrict__ gates,
                                                    float* __restrict__ hs, float* __restrict__ cs,
                                                    _Float16* __restrict__ xh,
                                                    const float* __restrict__ pos,
                                                    const float* __restrict__ Wout,
                                                    const float* __restrict__ bout,
                                                    float* __restrict__ nxt) {
  __shared__ float r0[HDIM], r1[HDIM];
  int n = blockIdx.x, h = threadIdx.x;
  const float* g = gates + (size_t)n * NGATE;
  float ig = sigmoidf_(g[h]);
  float fg = sigmoidf_(g[HDIM + h]);
  float gg = tanhf(g[2 * HDIM + h]);
  float og = sigmoidf_(g[3 * HDIM + h]);
  float c2 = fg * cs[n * HDIM + h] + ig * gg;
  float h2 = og * tanhf(c2);
  cs[n * HDIM + h] = c2;
  hs[n * HDIM + h] = h2;
  xh[(size_t)n * XDIM + (EDIM + PDIM) + h] = (_Float16)h2;  // h feeds next step's gate GEMM
  r0[h] = h2 * Wout[h];
  r1[h] = h2 * Wout[HDIM + h];
  __syncthreads();
  for (int s = HDIM / 2; s > 0; s >>= 1) {
    if (h < s) { r0[h] += r0[h + s]; r1[h] += r1[h + s]; }
    __syncthreads();
  }
  if (h == 0) {
    nxt[2 * n]     = pos[2 * n]     + r0[0] + bout[0];
    nxt[2 * n + 1] = pos[2 * n + 1] + r1[0] + bout[1];
  }
}

// --------------------------------------------------------------- launcher ----
extern "C" void kernel_launch(void* const* d_in, const int* in_sizes, int n_in,
                              void* d_out, int out_size, void* d_ws, size_t ws_size,
                              hipStream_t stream) {
  const float* observed = (const float*)d_in[0];
  const float* W_pos   = (const float*)d_in[1];
  const float* b_pos   = (const float*)d_in[2];
  const float* W_pool  = (const float*)d_in[3];
  const float* b_pool  = (const float*)d_in[4];
  const float* W_ihE   = (const float*)d_in[5];
  const float* b_ihE   = (const float*)d_in[6];
  const float* W_hhE   = (const float*)d_in[7];
  const float* b_hhE   = (const float*)d_in[8];
  const float* W_ihD   = (const float*)d_in[9];
  const float* b_ihD   = (const float*)d_in[10];
  const float* W_hhD   = (const float*)d_in[11];
  const float* b_hhD   = (const float*)d_in[12];
  const float* W_out   = (const float*)d_in[13];
  const float* b_out   = (const float*)d_in[14];
  float* out = (float*)d_out;
  int n_predict = out_size / (NAGENT * 2);   // = 12 (n_predict scalar lives on device)

  char* w = (char*)d_ws;
  size_t off = 0;
  auto carve = [&](size_t bytes) -> char* {
    char* p = w + off;
    off += (bytes + 255) & ~(size_t)255;
    return p;
  };
  float*    obs     = (float*)   carve((size_t)TOBS * NAGENT * 2 * sizeof(float));
  _Float16* Wpool16 = (_Float16*)carve((size_t)PDIM * KPOOL * sizeof(_Float16));
  _Float16* WcatE   = (_Float16*)carve((size_t)NGATE * XDIM * sizeof(_Float16));
  _Float16* WcatD   = (_Float16*)carve((size_t)NGATE * XDIM * sizeof(_Float16));
  float*    bgE     = (float*)   carve(NGATE * sizeof(float));
  float*    bgD     = (float*)   carve(NGATE * sizeof(float));
  float*    hs      = (float*)   carve((size_t)NAGENT * HDIM * sizeof(float));
  float*    cs      = (float*)   carve((size_t)NAGENT * HDIM * sizeof(float));
  _Float16* pooled  = (_Float16*)carve((size_t)NAGENT * KPOOL * sizeof(_Float16));
  _Float16* xh      = (_Float16*)carve((size_t)NAGENT * XDIM * sizeof(_Float16));
  float*    gates   = (float*)   carve((size_t)NAGENT * NGATE * sizeof(float));
  float*    dummy   = (float*)   carve(NAGENT * 2 * sizeof(float));

  impute_kernel<<<2, 256, 0, stream>>>(observed, obs);
  prep_kernel<<<1024, 256, 0, stream>>>(W_pool, W_ihE, W_hhE, W_ihD, W_hhD,
                                        b_ihE, b_hhE, b_ihD, b_hhD,
                                        Wpool16, WcatE, WcatD, bgE, bgD, hs, cs, xh);

  auto do_step = [&](const float* pos, const float* prev, const _Float16* Wcat,
                     const float* bg, float* nxt) {
    pool_kernel<<<NAGENT, HDIM, 0, stream>>>(pos, hs, pooled);
    socgemm_kernel<<<64, 128, 0, stream>>>(pooled, Wpool16, b_pool, xh);
    emb_kernel<<<(NAGENT * EDIM) / 256, 256, 0, stream>>>(pos, prev, W_pos, b_pos, xh);
    gatesgemm_kernel<<<128, 256, 0, stream>>>(xh, Wcat, bg, gates);
    lstm_kernel<<<NAGENT, HDIM, 0, stream>>>(gates, hs, cs, xh, pos, W_out, b_out, nxt);
  };

  // encoder: t = 1..7, output position discarded
  for (int t = 1; t < TOBS; t++)
    do_step(obs + (size_t)t * NAGENT * 2, obs + (size_t)(t - 1) * NAGENT * 2, WcatE, bgE, dummy);

  // decoder: predictions chained through d_out slices
  const float* obs7 = obs + (size_t)(TOBS - 1) * NAGENT * 2;
  for (int s = 0; s < n_predict; s++) {
    const float* curr = (s == 0) ? obs7 : out + (size_t)(s - 1) * NAGENT * 2;
    const float* prev = (s <= 1) ? obs7 : out + (size_t)(s - 2) * NAGENT * 2;
    do_step(curr, prev, WcatD, bgD, out + (size_t)s * NAGENT * 2);
  }
  (void)in_sizes; (void)n_in; (void)ws_size;
}
